// BiMamba_67731634257995
// MI455X (gfx1250) — matrix operand order repaired and checked by reading, not compiled, not run on hardware
//
#include <hip/hip_runtime.h>

// ---------------- problem constants ----------------
static constexpr int BATCH   = 4;
static constexpr int SEQ     = 4096;
static constexpr int DMODEL  = 768;
static constexpr int DSTATE  = 64;
static constexpr int DINNER  = 1536;
static constexpr int HEADDIM = 64;
static constexpr int NH      = 24;          // heads
static constexpr int CONVD   = 1664;        // DINNER + 2*DSTATE
static constexpr int DIP     = 3224;        // in_proj out features
static constexpr int LDZ     = 3264;        // padded DIP (51*64)
static constexpr int CHUNK   = 64;
static constexpr int NC      = SEQ / CHUNK; // 64
static constexpr int LP      = 72;          // LDS row pitch (ushorts), 144B: 16B-aligned rows
static constexpr size_t ML   = (size_t)BATCH * SEQ; // 16384 rows

// ---------------- types ----------------
typedef __attribute__((ext_vector_type(16))) __bf16 v16bf;
typedef __attribute__((ext_vector_type(8)))  float  v8f;
typedef __attribute__((ext_vector_type(4)))  unsigned int u32x4;

__device__ inline unsigned short f2bf(float f) {
  unsigned u = __builtin_bit_cast(unsigned, f);
  u += 0x7fffu + ((u >> 16) & 1u);           // round-to-nearest-even
  return (unsigned short)(u >> 16);
}
__device__ inline float bf2f(unsigned short h) {
  unsigned u = ((unsigned)h) << 16;
  return __builtin_bit_cast(float, u);
}
__device__ inline float siluf(float x) { return x / (1.f + __expf(-x)); }

// A-fragment: elements 0..7 at p[0..7], 8..15 at p[16..23]  (16B-aligned p)
__device__ inline v16bf ld_frag_a(const unsigned short* p) {
  union { v16bf bf; u32x4 q[2]; } f;
  f.q[0] = *(const u32x4*)p;
  f.q[1] = *(const u32x4*)(p + 16);
  return f.bf;
}
// B-fragment: 16 contiguous bf16 at p (16B-aligned)
__device__ inline v16bf ld_frag32(const unsigned short* p) {
  union { v16bf bf; u32x4 q[2]; } f;
  f.q[0] = *(const u32x4*)p;
  f.q[1] = *(const u32x4*)(p + 8);
  return f.bf;
}

// ---------------- elementwise converts ----------------
__global__ void k_cvt_bf16(const float* __restrict__ src,
                           unsigned short* __restrict__ dst, size_t n) {
  size_t i = (size_t)blockIdx.x * blockDim.x + threadIdx.x;
  size_t st = (size_t)gridDim.x * blockDim.x;
  for (; i < n; i += st) dst[i] = f2bf(src[i]);
}
__global__ void k_cvt_pad(const float* __restrict__ src,
                          unsigned short* __restrict__ dst,
                          int rows, int padRows, int cols) {
  size_t n = (size_t)padRows * cols;
  size_t st = (size_t)gridDim.x * blockDim.x;
  for (size_t i = (size_t)blockIdx.x * blockDim.x + threadIdx.x; i < n; i += st) {
    int rr = (int)(i / cols);
    dst[i] = (rr < rows) ? f2bf(src[i]) : (unsigned short)0;
  }
}

// ---------------- WMMA GEMM: C[M,N] = A[M,K] * W[N,K]^T ----------------
// block = 256 threads (8 waves); wave computes a 16x64 strip; grid(N/64, M/128)
// W k-slice DMA'd to LDS via GLOBAL_LOAD_ASYNC_TO_LDS_B128 (ASYNCcnt path):
// 256 lanes x 16B = full 64x32 bf16 tile per k-step, shared by all 8 waves.
// MODE 0: fp32 out, 1: bf16 out, 2: fp32 + bias
template <int MODE>
__global__ void k_gemm(const unsigned short* __restrict__ A,
                       const unsigned short* __restrict__ W,
                       void* __restrict__ Cout, const float* __restrict__ bias,
                       int K, int lda, int ldc) {
  const int tid = threadIdx.x;
  const int lane = tid & 31, wv = tid >> 5;
  const int half = lane >> 4, r = lane & 15;
  const int m0 = blockIdx.y * 128 + wv * 16;
  const int n0 = blockIdx.x * 64;
  __shared__ alignas(16) unsigned short sW[64 * 32];
  v8f acc[4] = {};
  const unsigned short* arow = A + (size_t)(m0 + r) * lda;
  const int sn = tid >> 2;              // 0..63 staging row
  const int sk = (tid & 3) * 8;         // 0,8,16,24
  const unsigned short* wstage = W + (size_t)(n0 + sn) * K + sk;
  const unsigned ldsDst = (unsigned)(size_t)&sW[sn * 32 + sk]; // LDS byte offset
  for (int k0 = 0; k0 < K; k0 += 32) {
    __syncthreads();                    // previous tile consumers done
    {
      unsigned long long ga = (unsigned long long)(size_t)(wstage + k0);
      asm volatile("global_load_async_to_lds_b128 %0, %1, off"
                   :: "v"(ldsDst), "v"(ga) : "memory");
      asm volatile("s_wait_asynccnt 0x0" ::: "memory");
    }
    __syncthreads();                    // all waves' DMA complete
    const unsigned short* ap = arow + k0 + 8 * half;
    __builtin_prefetch(ap + 32, 0, 1);          // next k-step A
    __builtin_prefetch(wstage + k0 + 32, 0, 1); // next k-step W
    v16bf a = ld_frag_a(ap);
#pragma unroll
    for (int tt = 0; tt < 4; ++tt) {
      v16bf bb = ld_frag_a(&sW[(tt * 16 + r) * 32 + 8 * half]);
      acc[tt] = __builtin_amdgcn_wmma_f32_16x16x32_bf16(false, a, false, bb,
                                                        (short)0, acc[tt], false, false);
    }
  }
#pragma unroll
  for (int tt = 0; tt < 4; ++tt)
#pragma unroll
    for (int v = 0; v < 8; ++v) {
      int m = m0 + v + 8 * half;
      int n = n0 + tt * 16 + r;
      float x = acc[tt][v];
      if (MODE == 0)      ((float*)Cout)[(size_t)m * ldc + n] = x;
      else if (MODE == 1) ((unsigned short*)Cout)[(size_t)m * ldc + n] = f2bf(x);
      else                ((float*)Cout)[(size_t)m * ldc + n] = x + bias[n];
    }
}

// ---------------- conv4 + SiLU + dt/softplus prep ----------------
__global__ void k_conv_prep(const float* __restrict__ Z,
                            const float* __restrict__ conv_w,
                            const float* __restrict__ conv_b,
                            const float* __restrict__ dt_bias,
                            const float* __restrict__ A_log,
                            unsigned short* __restrict__ xs,
                            unsigned short* __restrict__ dtx,
                            unsigned short* __restrict__ Bc,
                            unsigned short* __restrict__ Cc,
                            float* __restrict__ ldA, int rev) {
  const int bl = blockIdx.x;
  const int b = bl / SEQ, l = bl % SEQ;
  const int to = rev ? (SEQ - 1 - l) : l;
  const int tid = threadIdx.x;
  __shared__ float dts[NH];
  if (tid < NH) {
    float dtv = Z[((size_t)b * SEQ + to) * LDZ + 3200 + tid] + dt_bias[tid];
    float sp = dtv > 20.f ? dtv : log1pf(__expf(dtv));
    dts[tid] = sp;
    ldA[(size_t)bl * NH + tid] = -__expf(A_log[tid]) * sp;
  }
  __syncthreads();
  for (int c = tid; c < CONVD; c += blockDim.x) {
    float acc = conv_b[c];
#pragma unroll
    for (int j = 0; j < 4; ++j) {
      int ts = rev ? (to + 3 - j) : (to - 3 + j);
      if (ts >= 0 && ts < SEQ)
        acc += conv_w[c * 4 + j] * Z[((size_t)b * SEQ + ts) * LDZ + 1536 + c];
    }
    float v = siluf(acc);
    if (c < DINNER) {
      int h = c >> 6;
      xs[(size_t)bl * DINNER + c] = f2bf(v);
      dtx[(size_t)bl * DINNER + c] = f2bf(v * dts[h]);
    } else if (c < DINNER + DSTATE) {
      Bc[(size_t)bl * DSTATE + (c - DINNER)] = f2bf(v);
    } else {
      Cc[(size_t)bl * DSTATE + (c - DINNER - DSTATE)] = f2bf(v);
    }
  }
}

// ---------------- inclusive cumsum of log(dA) per (b,h,chunk) ----------------
__global__ void k_cumsum(const float* __restrict__ ldA, float* __restrict__ cs) {
  int idx = blockIdx.x;              // (b*NH + h)*NC + c
  int c = idx % NC; int h = (idx / NC) % NH; int b = idx / (NC * NH);
  int t = threadIdx.x;               // 64 threads
  __shared__ float s[64];
  s[t] = ldA[((size_t)b * SEQ + (size_t)c * CHUNK + t) * NH + h];
  __syncthreads();
  for (int off = 1; off < 64; off <<= 1) {
    float v = (t >= off) ? s[t - off] : 0.f;
    __syncthreads();
    s[t] += v;
    __syncthreads();
  }
  cs[(size_t)idx * 64 + t] = s[t];
}

// ---------------- per-chunk state:  S[n,p] = sum_s B[s,n]*e^{T-cs_s}*dtx[s,p] --
// LDS holds TRANSPOSED operands so every fragment read is contiguous (b128).
__global__ void k_chunk_state(const unsigned short* __restrict__ Bc,
                              const unsigned short* __restrict__ dtx,
                              const float* __restrict__ cs,
                              float* __restrict__ S) {
  const int c = blockIdx.x, h = blockIdx.y, b = blockIdx.z;
  const int tid = threadIdx.x;       // 128 = 4 waves
  __shared__ alignas(16) unsigned short sBT[64 * LP]; // [n][s]  (scaled B^T)
  __shared__ alignas(16) unsigned short sXT[64 * LP]; // [p][s]  (dtx^T)
  __shared__ float w[64];
  const size_t csb = (((size_t)b * NH + h) * NC + c) * 64;
  if (tid < 64) w[tid] = __expf(cs[csb + 63] - cs[csb + tid]);
  __syncthreads();
  const size_t blb = (size_t)b * SEQ + (size_t)c * CHUNK;
  for (int i = tid; i < 4096; i += 128) {
    int s = i >> 6, n = i & 63;      // coalesced global reads, scattered LDS writes
    sBT[n * LP + s] = f2bf(bf2f(Bc[(blb + s) * DSTATE + n]) * w[s]);
    sXT[n * LP + s] = dtx[(blb + s) * DINNER + h * HEADDIM + n];
  }
  __syncthreads();
  const int wv = tid >> 5, lane = tid & 31, half = lane >> 4, r = lane & 15;
  const int n0 = wv * 16;
  v8f acc[4] = {};
#pragma unroll
  for (int kk = 0; kk < 64; kk += 32) {
    v16bf a = ld_frag_a(&sBT[(n0 + r) * LP + kk + 8 * half]);
#pragma unroll
    for (int tt = 0; tt < 4; ++tt) {
      v16bf bb = ld_frag_a(&sXT[(tt * 16 + r) * LP + kk + 8 * half]);
      acc[tt] = __builtin_amdgcn_wmma_f32_16x16x32_bf16(false, a, false, bb,
                                                        (short)0, acc[tt], false, false);
    }
  }
  float* So = S + (((size_t)b * NH + h) * NC + c) * 4096;
#pragma unroll
  for (int tt = 0; tt < 4; ++tt)
#pragma unroll
    for (int v = 0; v < 8; ++v)
      So[(n0 + v + 8 * half) * 64 + (tt * 16 + r)] = acc[tt][v];
}

// ---------------- chunk-level prefix scan (in place -> S_prev per chunk) -----
__global__ void k_state_scan(float* __restrict__ S, const float* __restrict__ cs) {
  const int h = blockIdx.x % NH, b = blockIdx.x / NH;
  const int tid = threadIdx.x;       // 256
  float run[16];
#pragma unroll
  for (int i = 0; i < 16; ++i) run[i] = 0.f;
  const size_t base = ((size_t)b * NH + h) * NC * 4096;
  const size_t csb = (((size_t)b * NH + h) * NC) * 64;
  for (int k = 0; k < NC; ++k) {
    float dk = __expf(cs[csb + (size_t)k * 64 + 63]);
    float* p = S + base + (size_t)k * 4096;
#pragma unroll
    for (int i = 0; i < 16; ++i) {
      int e = tid + 256 * i;
      float v = p[e];
      p[e] = run[i];                 // state entering chunk k
      run[i] = dk * run[i] + v;
    }
  }
}

// ---------------- per-chunk output ----------------
// Y = (L ⊙ CBᵀ)·dtx  +  diag(e^cs)·(C·S_prev)  +  D·xs
__global__ void k_chunk_y(const unsigned short* __restrict__ Cc,
                          const unsigned short* __restrict__ Bc,
                          const unsigned short* __restrict__ dtx,
                          const unsigned short* __restrict__ xs,
                          const float* __restrict__ S,
                          const float* __restrict__ cs,
                          const float* __restrict__ Dp,
                          float* __restrict__ Y) {
  const int c = blockIdx.x, h = blockIdx.y, b = blockIdx.z;
  const int tid = threadIdx.x;       // 128 = 4 waves
  __shared__ alignas(16) unsigned short sC[64 * 64];   // [t][n]   row-major
  __shared__ alignas(16) unsigned short sB[64 * 64];   // [s][n]   row-major
  __shared__ alignas(16) unsigned short sG[64 * 64];   // [t][s]   masked CBᵀ
  __shared__ alignas(16) unsigned short sXT[64 * LP];  // [p][s]   dtx^T
  __shared__ alignas(16) unsigned short sST[64 * LP];  // [p][n]   S_prev^T
  __shared__ float csl[64];
  const size_t csb = (((size_t)b * NH + h) * NC + c) * 64;
  if (tid < 64) csl[tid] = cs[csb + tid];
  const size_t blb = (size_t)b * SEQ + (size_t)c * CHUNK;
  const float* Sp = S + (((size_t)b * NH + h) * NC + c) * 4096;
  for (int i = tid; i < 4096; i += 128) {
    int s = i >> 6, n = i & 63;
    sC[i] = Cc[(blb + s) * DSTATE + n];
    sB[i] = Bc[(blb + s) * DSTATE + n];
    sXT[n * LP + s] = dtx[(blb + s) * DINNER + h * HEADDIM + n];
    sST[n * LP + s] = f2bf(Sp[i]);   // here s=state row, n=headdim col
  }
  __syncthreads();
  const int wv = tid >> 5, lane = tid & 31, half = lane >> 4, r = lane & 15;
  const int t0 = wv * 16;
  const int trow = t0 + r;
  // pass 1: G = C · Bᵀ   (K = d_state; both operands contiguous in LDS)
  v8f g[4] = {};
#pragma unroll
  for (int kk = 0; kk < 64; kk += 32) {
    v16bf a = ld_frag_a(&sC[trow * 64 + kk + 8 * half]);
#pragma unroll
    for (int tt = 0; tt < 4; ++tt) {
      v16bf bb = ld_frag_a(&sB[(tt * 16 + r) * 64 + kk + 8 * half]);
      g[tt] = __builtin_amdgcn_wmma_f32_16x16x32_bf16(false, a, false, bb,
                                                      (short)0, g[tt], false, false);
    }
  }
  // causal decay mask: Ĝ[t,s] = G * e^{cs_t - cs_s}, s<=t
#pragma unroll
  for (int tt = 0; tt < 4; ++tt)
#pragma unroll
    for (int v = 0; v < 8; ++v) {
      int t = t0 + v + 8 * half;
      int s = tt * 16 + r;
      float val = (s <= t) ? g[tt][v] * __expf(csl[t] - csl[s]) : 0.f;
      sG[t * 64 + s] = f2bf(val);
    }
  __syncthreads();
  // pass 2: y  = Ĝ·dtx      (K = time)
  // pass 3: y2 = C·S_prev   (K = state), scaled by e^{cs_t} at store
  v8f y[4] = {}, y2[4] = {};
#pragma unroll
  for (int kk = 0; kk < 64; kk += 32) {
    v16bf a1 = ld_frag_a(&sG[trow * 64 + kk + 8 * half]);
    v16bf a2 = ld_frag_a(&sC[trow * 64 + kk + 8 * half]);
#pragma unroll
    for (int tt = 0; tt < 4; ++tt) {
      int p = tt * 16 + r;
      v16bf b1 = ld_frag_a(&sXT[p * LP + kk + 8 * half]);
      v16bf b2 = ld_frag_a(&sST[p * LP + kk + 8 * half]);
      y[tt]  = __builtin_amdgcn_wmma_f32_16x16x32_bf16(false, a1, false, b1,
                                                       (short)0, y[tt], false, false);
      y2[tt] = __builtin_amdgcn_wmma_f32_16x16x32_bf16(false, a2, false, b2,
                                                       (short)0, y2[tt], false, false);
    }
  }
  const float Dh = Dp[h];
#pragma unroll
  for (int tt = 0; tt < 4; ++tt)
#pragma unroll
    for (int v = 0; v < 8; ++v) {
      int t = t0 + v + 8 * half;
      int p = tt * 16 + r;
      size_t row = blb + t;
      float xv = bf2f(xs[row * DINNER + h * HEADDIM + p]);
      Y[row * DINNER + h * HEADDIM + p] =
          y[tt][v] + __expf(csl[t]) * y2[tt][v] + Dh * xv;
    }
}

// ---------------- gate (silu(z)) + RMS norm -> bf16 row in ORIGINAL order ----
__global__ void k_gate_norm(const float* __restrict__ Y, const float* __restrict__ Z,
                            const float* __restrict__ norm_w,
                            unsigned short* __restrict__ ybf, int rev) {
  const int bl = blockIdx.x;
  const int b = bl / SEQ, l = bl % SEQ;
  const int to = rev ? (SEQ - 1 - l) : l;
  const float* yrow = Y + (size_t)bl * DINNER;
  const float* zrow = Z + ((size_t)b * SEQ + to) * LDZ;   // z = first 1536 cols
  const int tid = threadIdx.x;      // 256
  float vals[6]; float ss = 0.f;
#pragma unroll
  for (int i = 0; i < 6; ++i) {
    int f = tid + 256 * i;
    float g = yrow[f] * siluf(zrow[f]);
    vals[i] = g; ss += g * g;
  }
  __shared__ float red[256];
  red[tid] = ss; __syncthreads();
  for (int o = 128; o > 0; o >>= 1) { if (tid < o) red[tid] += red[tid + o]; __syncthreads(); }
  float scale = rsqrtf(red[0] / (float)DINNER + 1e-5f);
  size_t ro = (size_t)b * SEQ + to;
#pragma unroll
  for (int i = 0; i < 6; ++i) {
    int f = tid + 256 * i;
    ybf[ro * DINNER + f] = f2bf(vals[i] * scale * norm_w[f]);
  }
}

// ---------------- workspace layout ----------------
static constexpr size_t au(size_t x) { return (x + 255) & ~(size_t)255; }
static constexpr size_t OFF_XBF  = 0;
static constexpr size_t OFF_WIN  = OFF_XBF  + au(ML * DMODEL * 2);
static constexpr size_t OFF_WOUT = OFF_WIN  + au((size_t)LDZ * DMODEL * 2);
static constexpr size_t OFF_WFUS = OFF_WOUT + au((size_t)DMODEL * DINNER * 2);
static constexpr size_t OFF_Z    = OFF_WFUS + au((size_t)DMODEL * DINNER * 2);
static constexpr size_t OFF_XS   = OFF_Z    + au(ML * LDZ * 4);
static constexpr size_t OFF_DTX  = OFF_XS   + au(ML * DINNER * 2);
static constexpr size_t OFF_BC   = OFF_DTX  + au(ML * DINNER * 2);
static constexpr size_t OFF_CC   = OFF_BC   + au(ML * DSTATE * 2);
static constexpr size_t OFF_LDA  = OFF_CC   + au(ML * DSTATE * 2);
static constexpr size_t OFF_CS   = OFF_LDA  + au(ML * NH * 4);
static constexpr size_t OFF_S    = OFF_CS   + au((size_t)BATCH * NH * NC * 64 * 4);
static constexpr size_t OFF_Y    = OFF_S    + au((size_t)BATCH * NH * NC * 4096 * 4);
static constexpr size_t OFF_YBF  = OFF_Y    + au(ML * DINNER * 4);
static constexpr size_t OFF_CAT  = OFF_YBF  + au(ML * DINNER * 2);

extern "C" void kernel_launch(void* const* d_in, const int* in_sizes, int n_in,
                              void* d_out, int out_size, void* d_ws, size_t ws_size,
                              hipStream_t stream) {
  (void)in_sizes; (void)n_in; (void)out_size; (void)ws_size;
  char* ws = (char*)d_ws;
  unsigned short* xbf  = (unsigned short*)(ws + OFF_XBF);
  unsigned short* wIn  = (unsigned short*)(ws + OFF_WIN);
  unsigned short* wOut = (unsigned short*)(ws + OFF_WOUT);
  unsigned short* wFus = (unsigned short*)(ws + OFF_WFUS);
  float*          Z    = (float*)(ws + OFF_Z);
  unsigned short* xs   = (unsigned short*)(ws + OFF_XS);
  unsigned short* dtx  = (unsigned short*)(ws + OFF_DTX);
  unsigned short* Bc   = (unsigned short*)(ws + OFF_BC);
  unsigned short* Cc   = (unsigned short*)(ws + OFF_CC);
  float*          ldA  = (float*)(ws + OFF_LDA);
  float*          cs   = (float*)(ws + OFF_CS);
  float*          S    = (float*)(ws + OFF_S);
  float*          Y    = (float*)(ws + OFF_Y);
  unsigned short* ybf  = (unsigned short*)(ws + OFF_YBF);
  unsigned short* cat  = (unsigned short*)(ws + OFF_CAT);

  const float* x     = (const float*)d_in[0];
  const float* fus_w = (const float*)d_in[1];
  const float* fus_b = (const float*)d_in[2];

  // shared conversions
  k_cvt_bf16<<<4096, 256, 0, stream>>>(x, xbf, ML * DMODEL);
  k_cvt_bf16<<<1024, 256, 0, stream>>>(fus_w, wFus, (size_t)DMODEL * DINNER);

  for (int dir = 0; dir < 2; ++dir) {
    const int base = dir ? 11 : 3;
    const float* in_w    = (const float*)d_in[base + 0];
    const float* conv_w  = (const float*)d_in[base + 1];
    const float* conv_b  = (const float*)d_in[base + 2];
    const float* dt_bias = (const float*)d_in[base + 3];
    const float* A_log   = (const float*)d_in[base + 4];
    const float* Dp      = (const float*)d_in[base + 5];
    const float* norm_w  = (const float*)d_in[base + 6];
    const float* out_w   = (const float*)d_in[base + 7];

    k_cvt_pad<<<2048, 256, 0, stream>>>(in_w, wIn, DIP, LDZ, DMODEL);
    k_cvt_bf16<<<1024, 256, 0, stream>>>(out_w, wOut, (size_t)DMODEL * DINNER);

    // in-proj GEMM: Z[16384, 3264] = xbf[16384,768] * wIn[3264,768]^T
    k_gemm<0><<<dim3(LDZ / 64, ML / 128), 256, 0, stream>>>(
        xbf, wIn, Z, nullptr, DMODEL, DMODEL, LDZ);

    k_conv_prep<<<(int)ML, 256, 0, stream>>>(Z, conv_w, conv_b, dt_bias, A_log,
                                             xs, dtx, Bc, Cc, ldA, dir);
    k_cumsum<<<BATCH * NH * NC, 64, 0, stream>>>(ldA, cs);
    k_chunk_state<<<dim3(NC, NH, BATCH), 128, 0, stream>>>(Bc, dtx, cs, S);
    k_state_scan<<<BATCH * NH, 256, 0, stream>>>(S, cs);
    k_chunk_y<<<dim3(NC, NH, BATCH), 128, 0, stream>>>(Cc, Bc, dtx, xs, S, cs, Dp, Y);
    k_gate_norm<<<(int)ML, 256, 0, stream>>>(Y, Z, norm_w, ybf, dir);

    // out-proj GEMM -> bf16 into concat buffer (f at col 0, b at col 768)
    k_gemm<1><<<dim3(DMODEL / 64, ML / 128), 256, 0, stream>>>(
        ybf, wOut, cat + (dir ? DMODEL : 0), nullptr, DINNER, DINNER, DINNER);
  }

  // fusion GEMM: out[16384,768] = cat[16384,1536] * fus_w[768,1536]^T + fus_b
  k_gemm<2><<<dim3(DMODEL / 64, ML / 128), 256, 0, stream>>>(
      cat, wFus, d_out, fus_b, DINNER, DINNER, DMODEL);
}
